// MultiheadAttention_87144886435875
// MI455X (gfx1250) — matrix operand order repaired and checked by reading, not compiled, not run on hardware
//
#include <hip/hip_runtime.h>

typedef __attribute__((ext_vector_type(16))) _Float16 v16h;
typedef __attribute__((ext_vector_type(8)))  float    v8f;

#define WMMA_F16(A, Bv, C) \
  __builtin_amdgcn_wmma_f32_16x16x32_f16(false, (A), false, (Bv), (short)0, (C), false, false)

constexpr int B_  = 4;
constexpr int S_  = 1024;
constexpr int D_  = 1024;
constexpr int H_  = 16;
constexpr int DK_ = 64;
constexpr int M_  = B_ * S_;           // 4096 rows for the projections
constexpr float NEGV = -1000000000.0f; // matches reference NEG

// ---------------- wave32 16-lane reductions (per half-wave row groups) ------
__device__ __forceinline__ float red_max16(float v) {
  #pragma unroll
  for (int d = 1; d < 16; d <<= 1) v = fmaxf(v, __shfl_xor(v, d, 32));
  return v;
}
__device__ __forceinline__ float red_sum16(float v) {
  #pragma unroll
  for (int d = 1; d < 16; d <<= 1) v += __shfl_xor(v, d, 32);
  return v;
}

// ---------------------------------------------------------------------------
// Projection GEMM: Y[m,n] = X[m,:] . W[n,:] + bias[n]    (X:[M,D], W:[D,D])
// One wave computes a 32x64 output block: per K-step (32), two A operands
// share four B operands -> 8 WMMAs per 4 B fetches (halved weight traffic).
// mode 0: fp32 row-major out (final output projection)
// mode 1: f16 out, per-head layout  [B,H,S,DK]   (Q and K)
// mode 2: f16 out, per-head layout  [B,H,DK,S]   (V transposed for PV GEMM)
// ---------------------------------------------------------------------------
__global__ __launch_bounds__(32)
void proj_kernel(const float* __restrict__ X, const float* __restrict__ W,
                 const float* __restrict__ bias,
                 float* __restrict__ outF, _Float16* __restrict__ outH, int mode)
{
  const int m0   = blockIdx.x * 32;
  const int n0   = blockIdx.y * 64;
  const int lane = threadIdx.x;
  const int half = lane >> 4;   // 0: lanes 0-15, 1: lanes 16-31
  const int l15  = lane & 15;

  v8f acc[2][4] = {};
  const float* Xr0 = X + (size_t)(m0 + l15) * D_;
  const float* Xr1 = X + (size_t)(m0 + 16 + l15) * D_;

  for (int k0 = 0; k0 < D_; k0 += 32) {
    // A operands (16x32 f16): lane half selects interleaved K blocks of 8
    v16h A0, A1;
    const int ab = k0 + half * 8;
    #pragma unroll
    for (int j = 0; j < 8; ++j) {
      A0[j]     = (_Float16)Xr0[ab + j];
      A0[8 + j] = (_Float16)Xr0[ab + 16 + j];
      A1[j]     = (_Float16)Xr1[ab + j];
      A1[8 + j] = (_Float16)Xr1[ab + 16 + j];
    }
    #pragma unroll
    for (int t = 0; t < 4; ++t) {
      // B operand (32x16): lane l15 holds column n = row (n0+t*16+l15) of W
      const float* Wr = W + (size_t)(n0 + t * 16 + l15) * D_ + k0 + half * 8;
      v16h Bv;
      #pragma unroll
      for (int j = 0; j < 8; ++j) { Bv[j] = (_Float16)Wr[j]; Bv[8 + j] = (_Float16)Wr[16 + j]; }
      acc[0][t] = WMMA_F16(A0, Bv, acc[0][t]);
      acc[1][t] = WMMA_F16(A1, Bv, acc[1][t]);
    }
  }

  #pragma unroll
  for (int a = 0; a < 2; ++a) {
    #pragma unroll
    for (int t = 0; t < 4; ++t) {
      const int col = n0 + t * 16 + l15;
      const float bvv = bias[col];
      #pragma unroll
      for (int r = 0; r < 8; ++r) {
        const int row = m0 + a * 16 + r + half * 8;  // C layout: VGPR r -> M=r / M=8+r
        const float v = acc[a][t][r] + bvv;
        if (mode == 0) {
          outF[(size_t)row * D_ + col] = v;
        } else {
          const int bb = row >> 10, ss = row & (S_ - 1);
          const int hh = col >> 6,  dd = col & 63;
          if (mode == 1)
            outH[(((size_t)(bb * H_ + hh)) * S_ + ss) * DK_ + dd] = (_Float16)v;
          else
            outH[(((size_t)(bb * H_ + hh)) * DK_ + dd) * S_ + ss] = (_Float16)v;
        }
      }
    }
  }
}

// ---------------------------------------------------------------------------
// Fused attention: for one (b,h) and one 16-row query tile, stream keys in
// 32-wide chunks with online softmax. Output = 0.5*softmax(S)V + 0.5*(mask*R)V
// ---------------------------------------------------------------------------
__global__ __launch_bounds__(32)
void attn_kernel(const _Float16* __restrict__ Qh, const _Float16* __restrict__ Kh,
                 const _Float16* __restrict__ Vt, const float* __restrict__ Rel,
                 const unsigned char* __restrict__ mask,
                 const unsigned char* __restrict__ pad,
                 float* __restrict__ Xout)
{
  const int bh   = blockIdx.y;
  const int b    = bh >> 4;        // / H_
  const int h    = bh & 15;
  const int q0   = blockIdx.x * 16;
  const int lane = threadIdx.x;
  const int half = lane >> 4;
  const int l15  = lane & 15;

  const _Float16* Qb = Qh + (size_t)bh * S_ * DK_;
  const _Float16* Kb = Kh + (size_t)bh * S_ * DK_;
  const _Float16* Vb = Vt + (size_t)bh * DK_ * S_;
  const float*    Rb = Rel + (size_t)bh * S_ * S_;
  const unsigned char* Mb = mask + (size_t)b * S_ * S_;   // [S,S] (head-bcast)
  const unsigned char* Pb = pad  + (size_t)b * S_;

  // Q tile as two A operands (d = 0..31 and 32..63)
  v16h qA[2];
  {
    const _Float16* qr = Qb + (size_t)(q0 + l15) * DK_;
    #pragma unroll
    for (int c = 0; c < 2; ++c) {
      const int dbase = c * 32 + half * 8;
      #pragma unroll
      for (int j = 0; j < 8; ++j) {
        qA[c][j]     = qr[dbase + j];
        qA[c][8 + j] = qr[dbase + 16 + j];
      }
    }
  }

  v8f accS[4] = {};           // softmax numerator accumulator (rescaled)
  v8f accR[4] = {};           // relation-matrix term (exact)
  float mrow[8], lsum[8];
  #pragma unroll
  for (int r = 0; r < 8; ++r) { mrow[r] = -INFINITY; lsum[r] = 0.0f; }

  __shared__ _Float16 Ps[16 * 32];   // P (softmax part), C-layout -> A-layout bounce
  __shared__ _Float16 Pr[16 * 32];   // masked relation tile

  for (int k0 = 0; k0 < S_; k0 += 32) {
    // ---- scores: S[16 q, 32 k] as two 16x16 tiles, contraction over d=64 ----
    v8f s[2];
    #pragma unroll
    for (int nt = 0; nt < 2; ++nt) {
      v8f c = {};
      const _Float16* kr = Kb + (size_t)(k0 + nt * 16 + l15) * DK_;
      #pragma unroll
      for (int cs = 0; cs < 2; ++cs) {
        v16h Bv;
        const int dbase = cs * 32 + half * 8;
        #pragma unroll
        for (int j = 0; j < 8; ++j) { Bv[j] = kr[dbase + j]; Bv[8 + j] = kr[dbase + 16 + j]; }
        c = WMMA_F16(qA[cs], Bv, c);
      }
      s[nt] = c;
    }

    // ---- scale + mask (full_mask = mask[b,0,q,k] & padding_mask[b,k]) ----
    float mw[2][8];
    #pragma unroll
    for (int nt = 0; nt < 2; ++nt) {
      const int kcol = k0 + nt * 16 + l15;
      const bool pm = Pb[kcol] != 0;
      #pragma unroll
      for (int r = 0; r < 8; ++r) {
        const int qrow = q0 + r + half * 8;
        const bool fm = pm && (Mb[(size_t)qrow * S_ + kcol] != 0);
        mw[nt][r] = fm ? 1.0f : 0.0f;
        s[nt][r]  = fm ? s[nt][r] * 0.125f : NEGV;   // 1/sqrt(64) = 0.125
      }
    }

    // ---- online softmax: row max, rescale, exponentials ----
    float e[2][8];
    #pragma unroll
    for (int r = 0; r < 8; ++r) {
      float cand = red_max16(fmaxf(s[0][r], s[1][r]));
      const float mn = fmaxf(mrow[r], cand);
      const float alpha = __expf(mrow[r] - mn);      // 0 on first chunk
      mrow[r] = mn;
      lsum[r] *= alpha;
      #pragma unroll
      for (int dt = 0; dt < 4; ++dt) accS[dt][r] *= alpha;
      const float e0 = __expf(s[0][r] - mn);         // masked -> exp(-1e9)=0
      const float e1 = __expf(s[1][r] - mn);
      lsum[r] += e0 + e1;                            // per-lane partial row sum
      e[0][r] = e0; e[1][r] = e1;
    }

    // ---- C-layout -> A-layout transpose of P tiles through LDS ----
    #pragma unroll
    for (int nt = 0; nt < 2; ++nt) {
      const int colL = nt * 16 + l15;
      const int kcol = k0 + colL;
      #pragma unroll
      for (int r = 0; r < 8; ++r) {
        const int rowL = r + half * 8;
        const float fm = mw[nt][r];
        Ps[rowL * 32 + colL] = (_Float16)(e[nt][r] * fm);  // zero if masked
        const float rv = (fm != 0.0f)
            ? Rb[(size_t)(q0 + rowL) * S_ + kcol] : 0.0f;
        Pr[rowL * 32 + colL] = (_Float16)rv;
      }
    }
    __syncthreads();   // single-wave workgroup: lowers to S_NOP, waits handled

    v16h PsA, PrA;
    {
      const int kb = half * 8;
      const _Float16* psr = &Ps[l15 * 32];
      const _Float16* prr = &Pr[l15 * 32];
      #pragma unroll
      for (int j = 0; j < 8; ++j) {
        PsA[j]     = psr[kb + j];      PsA[8 + j] = psr[kb + 16 + j];
        PrA[j]     = prr[kb + j];      PrA[8 + j] = prr[kb + 16 + j];
      }
    }

    // ---- accumulate P @ V (V stored transposed: rows contiguous in k) ----
    #pragma unroll
    for (int dt = 0; dt < 4; ++dt) {
      const _Float16* vr = Vb + (size_t)(dt * 16 + l15) * S_ + k0 + half * 8;
      v16h Bv;
      #pragma unroll
      for (int j = 0; j < 8; ++j) { Bv[j] = vr[j]; Bv[8 + j] = vr[16 + j]; }
      accS[dt] = WMMA_F16(PsA, Bv, accS[dt]);
      accR[dt] = WMMA_F16(PrA, Bv, accR[dt]);
    }

    // prefetch next chunk of the (bandwidth-dominant) relation matrix
    if (k0 + 32 < S_)
      __builtin_prefetch(&Rb[(size_t)(q0 + half * 8) * S_ + k0 + 32 + l15 * 2], 0, 0);

    __syncthreads();   // Ps/Pr reused next iteration
  }

  // ---- finalize: x = 0.5 * accS / l + 0.5 * accR ----
  #pragma unroll
  for (int r = 0; r < 8; ++r) {
    const float L = red_sum16(lsum[r]);
    const float inv = (L > 0.0f) ? (1.0f / L) : 0.0f;  // all-masked row -> 0
    const int row = q0 + r + half * 8;
    #pragma unroll
    for (int dt = 0; dt < 4; ++dt) {
      const float val = 0.5f * accS[dt][r] * inv + 0.5f * accR[dt][r];
      Xout[((size_t)b * S_ + row) * D_ + h * DK_ + dt * 16 + l15] = val;
    }
  }
}

// ---------------------------------------------------------------------------
extern "C" void kernel_launch(void* const* d_in, const int* in_sizes, int n_in,
                              void* d_out, int out_size, void* d_ws, size_t ws_size,
                              hipStream_t stream)
{
  (void)in_sizes; (void)n_in; (void)out_size; (void)ws_size;

  const float* query = (const float*)d_in[0];
  const float* key_  = (const float*)d_in[1];
  const float* value = (const float*)d_in[2];
  const float* rel   = (const float*)d_in[3];
  const float* Wq    = (const float*)d_in[4];
  const float* bq    = (const float*)d_in[5];
  const float* Wk    = (const float*)d_in[6];
  const float* bk    = (const float*)d_in[7];
  const float* Wv    = (const float*)d_in[8];
  const float* bv    = (const float*)d_in[9];
  const float* Wo    = (const float*)d_in[10];
  const float* bo    = (const float*)d_in[11];
  const unsigned char* mask = (const unsigned char*)d_in[12]; // [B,1,S,S] bool
  const unsigned char* pad  = (const unsigned char*)d_in[13]; // [B,S] bool

  char* ws = (char*)d_ws;
  const size_t szH = (size_t)B_ * H_ * S_ * DK_ * sizeof(_Float16); // 8 MB each
  _Float16* Qh = (_Float16*)(ws);
  _Float16* Kh = (_Float16*)(ws + szH);
  _Float16* Vt = (_Float16*)(ws + 2 * szH);
  float*    Xb = (float*)   (ws + 3 * szH);                        // 16 MB fp32

  const dim3 blk(32);
  const dim3 pg(M_ / 32, D_ / 64);       // 128 x 16 waves, 32x64 tile each
  const dim3 ag(S_ / 16, B_ * H_);       // 64 x 64 waves
  (void)ag;

  proj_kernel<<<pg, blk, 0, stream>>>(query, Wq, bq, nullptr, Qh, 1);
  proj_kernel<<<pg, blk, 0, stream>>>(key_,  Wk, bk, nullptr, Kh, 1);
  proj_kernel<<<pg, blk, 0, stream>>>(value, Wv, bv, nullptr, Vt, 2);
  attn_kernel<<<dim3(S_ / 16, B_ * H_), blk, 0, stream>>>(Qh, Kh, Vt, rel, mask, pad, Xb);
  proj_kernel<<<pg, blk, 0, stream>>>(Xb, Wo, bo, (float*)d_out, nullptr, 0);
}
